// SelfAttention_74431783239945
// MI455X (gfx1250) — matrix operand order repaired and checked by reading, not compiled, not run on hardware
//
#include <hip/hip_runtime.h>

// ---------------------------------------------------------------------------
// LinearAttention (GroupNorm -> QKV -> linear attention -> out proj) for
// MI455X / gfx1250.  All heavy GEMMs use v_wmma_f32_16x16x32_bf16 (wave32).
// Re-associated einsums: S_h[c',c] = sum_n K[c',n]Q[c,n]  (32x32 per head),
// Weff[o,32h+c'] = sum_c w_out[o,32h+c] S_h[c',c],  y = Weff * V + b_out.
// ---------------------------------------------------------------------------

#define B_   16
#define C_   128
#define N_   1024
#define HID_ 384
#define SCALE_Q 0.17677669529663687f   // 32^-0.5

typedef __attribute__((ext_vector_type(16))) __bf16 v16bf;
typedef __attribute__((ext_vector_type(8)))  float  v8f;
typedef __attribute__((ext_vector_type(8)))  unsigned short us8;

union FragBF { v16bf v; us8 u[2]; };

__device__ __forceinline__ unsigned short f2bf(float f) {
  unsigned u = __float_as_uint(f);
  u += 0x7FFFu + ((u >> 16) & 1u);          // round-to-nearest-even
  return (unsigned short)(u >> 16);
}
__device__ __forceinline__ float bf2f(unsigned short h) {
  return __uint_as_float(((unsigned)h) << 16);
}

// A fragment (16x32 bf16) from row-major [M][K] tile at (m0,k0), leading dim ld.
// Layout used here (same k order as load_frag_b): lane L: M=L&15; half h=L>>4; elems 0..15 -> K = k0+h*16 .. +15
// (contiguous).  Two 16-byte loads per lane.
__device__ __forceinline__ v16bf load_frag_a(const unsigned short* p, int ld,
                                             int m0, int k0) {
  int L = threadIdx.x & 31, r = L & 15, h = L >> 4;
  const unsigned short* row = p + (size_t)(m0 + r) * ld + k0;
  FragBF f;
  f.u[0] = *(const us8*)(row + h * 16);
  f.u[1] = *(const us8*)(row + h * 16 + 8);
  return f.v;
}
// B fragment (32x16 bf16) from Bt stored row-major [N][K] at (n0,k0).
// Lane L: N=L&15; K = k0 + (L>>4)*16 + 0..15 (contiguous).
__device__ __forceinline__ v16bf load_frag_b(const unsigned short* p, int ld,
                                             int n0, int k0) {
  int L = threadIdx.x & 31, r = L & 15, h = L >> 4;
  const unsigned short* row = p + (size_t)(n0 + r) * ld + k0 + h * 16;
  FragBF f;
  f.u[0] = *(const us8*)(row);
  f.u[1] = *(const us8*)(row + 8);
  return f.v;
}
__device__ __forceinline__ v8f wmma_bf16(v16bf a, v16bf b, v8f c) {
  return __builtin_amdgcn_wmma_f32_16x16x32_bf16(false, a, false, b,
                                                 (short)0, c, false, false);
}

// ---------------------------------------------------------------- K1: GN stats
__global__ void k_gn_stats(const float* __restrict__ x, float* mu, float* rs) {
  int b = blockIdx.x, t = threadIdx.x;
  const float4* p = (const float4*)(x + (size_t)b * C_ * N_);
  float s = 0.f, s2 = 0.f;
  for (int i = t; i < C_ * N_ / 4; i += 256) {
    float4 v = p[i];
    s  += v.x + v.y + v.z + v.w;
    s2 += v.x * v.x + v.y * v.y + v.z * v.z + v.w * v.w;
  }
  __shared__ float ls[256], ls2[256];
  ls[t] = s; ls2[t] = s2; __syncthreads();
  for (int o = 128; o > 0; o >>= 1) {
    if (t < o) { ls[t] += ls[t + o]; ls2[t] += ls2[t + o]; }
    __syncthreads();
  }
  if (t == 0) {
    float m = ls[0] / (float)(C_ * N_);
    float var = ls2[0] / (float)(C_ * N_) - m * m;
    mu[b] = m;
    rs[b] = rsqrtf(var + 1e-5f);
  }
}

// ------------------------------------------------- K2: w_qkv fp32 -> bf16
__global__ void k_cvt_w(const float* __restrict__ wqkv, unsigned short* wqb) {
  int i = blockIdx.x * 256 + threadIdx.x;
  if (i < HID_ * C_) wqb[i] = f2bf(wqkv[i]);
}

// --------------------- K3: normalize + transpose-store xnT[b][n][c] (bf16)
__global__ void k_norm_t(const float* __restrict__ x, const float* __restrict__ gw,
                         const float* __restrict__ gb, const float* mu,
                         const float* rs, unsigned short* xnT) {
  int b = blockIdx.z, c0 = blockIdx.y * 32, n0 = blockIdx.x * 64;
  __shared__ float tile[32][65];              // +1 pad: conflict-free transpose
  float m = mu[b], r = rs[b];
  const float* xb = x + (size_t)b * C_ * N_;
#pragma unroll
  for (int i = 0; i < 8; i++) {
    int e = threadIdx.x + i * 256;
    int c = e >> 6, nn = e & 63;
    float v = xb[(size_t)(c0 + c) * N_ + n0 + nn];
    tile[c][nn] = (v - m) * r * gw[c0 + c] + gb[c0 + c];
  }
  __syncthreads();
  unsigned short* ob = xnT + (size_t)b * N_ * C_;
#pragma unroll
  for (int i = 0; i < 8; i++) {
    int e = threadIdx.x + i * 256;
    int nn = e >> 5, c = e & 31;
    ob[(size_t)(n0 + nn) * C_ + c0 + c] = f2bf(tile[c][nn]);
  }
}

// ------ K4: QKV GEMM.  D[o][n] = sum_c wq[o][c] * xn[c][n] per batch (WMMA).
// part 0/1 (q,k): natural bf16 store into qk[b][256][1024].
// part 2   (v) : transposed bf16 store into vt[b][n][c'] (ready as Bt).
__global__ void k_qkv_gemm(const unsigned short* __restrict__ wqb,
                           const unsigned short* __restrict__ xnT,
                           unsigned short* qk, unsigned short* vt) {
  int b = blockIdx.z, part = blockIdx.y;
  int wave = threadIdx.x >> 5;
  int o0 = part * 128 + wave * 16;
  int n0 = blockIdx.x * 64;
  const unsigned short* xb = xnT + (size_t)b * N_ * C_;
  v8f acc[4] = {};
  for (int kk = 0; kk < C_; kk += 32) {
    v16bf fa = load_frag_a(wqb, C_, o0, kk);
#pragma unroll
    for (int j = 0; j < 4; j++) {
      v16bf fb = load_frag_b(xb, C_, n0 + j * 16, kk);
      acc[j] = wmma_bf16(fa, fb, acc[j]);
    }
  }
  int L = threadIdx.x & 31, nL = L & 15, hh = L >> 4;
  if (part < 2) {
    unsigned short* outb = qk + (size_t)b * 256 * N_;
#pragma unroll
    for (int j = 0; j < 4; j++)
#pragma unroll
      for (int r = 0; r < 8; r++)
        outb[(size_t)(o0 + 8 * hh + r) * N_ + n0 + j * 16 + nL] = f2bf(acc[j][r]);
  } else {
    unsigned short* vb = vt + (size_t)b * N_ * C_;
    int m0 = o0 - 256;
#pragma unroll
    for (int j = 0; j < 4; j++) {
      unsigned short tmp[8];
#pragma unroll
      for (int r = 0; r < 8; r++) tmp[r] = f2bf(acc[j][r]);
      *(us8*)(vb + (size_t)(n0 + j * 16 + nL) * C_ + m0 + 8 * hh) = *(const us8*)tmp;
    }
  }
}

// -------------------- K5: q softmax over the 1024 tokens, in place, * SCALE
__global__ void k_smax_q(unsigned short* qk) {
  int o = blockIdx.x, b = blockIdx.y, t = threadIdx.x;
  unsigned short* row = qk + ((size_t)b * 256 + o) * N_;
  float v[4];
  float mx = -3.0e38f;
#pragma unroll
  for (int i = 0; i < 4; i++) { v[i] = bf2f(row[t + i * 256]); mx = fmaxf(mx, v[i]); }
  __shared__ float red[256];
  red[t] = mx; __syncthreads();
  for (int s = 128; s > 0; s >>= 1) {
    if (t < s) red[t] = fmaxf(red[t], red[t + s]);
    __syncthreads();
  }
  mx = red[0]; __syncthreads();
  float sum = 0.f;
#pragma unroll
  for (int i = 0; i < 4; i++) { v[i] = __expf(v[i] - mx); sum += v[i]; }
  red[t] = sum; __syncthreads();
  for (int s = 128; s > 0; s >>= 1) {
    if (t < s) red[t] += red[t + s];
    __syncthreads();
  }
  float inv = SCALE_Q / red[0];
#pragma unroll
  for (int i = 0; i < 4; i++) row[t + i * 256] = f2bf(v[i] * inv);
}

// -------------- K6: k softmax over 32 head-channels (per n), in place
__global__ void k_smax_k(unsigned short* qk) {
  int n = blockIdx.x * 256 + threadIdx.x;
  int h = blockIdx.y, b = blockIdx.z;
  unsigned short* base = qk + ((size_t)b * 256 + 128 + 32 * h) * N_ + n;
  float v[32];
  float mx = -3.0e38f;
#pragma unroll
  for (int j = 0; j < 32; j++) { v[j] = bf2f(base[(size_t)j * N_]); mx = fmaxf(mx, v[j]); }
  float sum = 0.f;
#pragma unroll
  for (int j = 0; j < 32; j++) { v[j] = __expf(v[j] - mx); sum += v[j]; }
  float inv = 1.f / sum;
#pragma unroll
  for (int j = 0; j < 32; j++) base[(size_t)j * N_] = f2bf(v[j] * inv);
}

// --- K7: S GEMM per (b,h): D[c][c'] = sum_n Qs[c][n]*Ks[c'][n] (K=1024, WMMA).
// Transposed store => S[b][h][c'][c] fp32, row-major over c.
__global__ void k_s_gemm(const unsigned short* __restrict__ qk, float* S) {
  int b = blockIdx.x >> 2, h = blockIdx.x & 3;
  int w = threadIdx.x >> 5;
  int m0 = (w & 1) * 16, n0 = (w >> 1) * 16;
  const unsigned short* qb = qk + ((size_t)b * 256 + 32 * h) * N_;
  const unsigned short* kb = qk + ((size_t)b * 256 + 128 + 32 * h) * N_;
  v8f acc = {};
  for (int kk = 0; kk < N_; kk += 32) {
    v16bf fa = load_frag_a(qb, N_, m0, kk);
    v16bf fb = load_frag_b(kb, N_, n0, kk);
    acc = wmma_bf16(fa, fb, acc);
  }
  int L = threadIdx.x & 31, nL = L & 15, hh = L >> 4;
  float* sb = S + (size_t)(b * 4 + h) * 32 * 32;
#pragma unroll
  for (int r = 0; r < 8; r++)
    sb[(size_t)(n0 + nL) * 32 + m0 + 8 * hh + r] = acc[r];
}

// ----- K8: Weff[b][o][32h+c'] = sum_c w_out[o][32h+c]*S[b][h][c'][c] (tiny)
__global__ void k_weff(const float* __restrict__ wout, const float* __restrict__ S,
                       unsigned short* weff) {
  int b = blockIdx.x >> 2, h = blockIdx.x & 3;
  __shared__ float sS[32][32];
  const float* sb = S + (size_t)(b * 4 + h) * 1024;
  for (int i = threadIdx.x; i < 1024; i += 256) sS[i >> 5][i & 31] = sb[i];
  __syncthreads();
  unsigned short* wb = weff + (size_t)b * C_ * C_;
  for (int i = threadIdx.x; i < 4096; i += 256) {
    int o = i >> 5, cp = i & 31;
    const float* wrow = wout + (size_t)o * C_ + 32 * h;
    float s = 0.f;
#pragma unroll
    for (int c = 0; c < 32; c++) s += wrow[c] * sS[cp][c];
    wb[(size_t)o * C_ + 32 * h + cp] = f2bf(s);
  }
}

// --- K9: y[b][o][n] = sum_c' Weff[b][o][c'] * Vt[b][n][c'] + b_out[o] (WMMA)
__global__ void k_final_gemm(const unsigned short* __restrict__ weff,
                             const unsigned short* __restrict__ vt,
                             const float* __restrict__ bout, float* y) {
  int b = blockIdx.z;
  int wave = threadIdx.x >> 5;
  int o0 = wave * 16;
  int n0 = blockIdx.x * 64;
  const unsigned short* wb = weff + (size_t)b * C_ * C_;
  const unsigned short* vb = vt + (size_t)b * N_ * C_;
  v8f acc[4] = {};
  for (int kk = 0; kk < C_; kk += 32) {
    v16bf fa = load_frag_a(wb, C_, o0, kk);
#pragma unroll
    for (int j = 0; j < 4; j++) {
      v16bf fb = load_frag_b(vb, C_, n0 + j * 16, kk);
      acc[j] = wmma_bf16(fa, fb, acc[j]);
    }
  }
  int L = threadIdx.x & 31, nL = L & 15, hh = L >> 4;
  float* yb = y + (size_t)b * C_ * N_;
#pragma unroll
  for (int j = 0; j < 4; j++)
#pragma unroll
    for (int r = 0; r < 8; r++) {
      int o = o0 + 8 * hh + r;
      yb[(size_t)o * N_ + n0 + j * 16 + nL] = acc[j][r] + bout[o];
    }
}

// ---------------------------------------------------------------------------
extern "C" void kernel_launch(void* const* d_in, const int* in_sizes, int n_in,
                              void* d_out, int out_size, void* d_ws, size_t ws_size,
                              hipStream_t stream) {
  const float* x    = (const float*)d_in[0];
  const float* gw   = (const float*)d_in[1];
  const float* gb   = (const float*)d_in[2];
  const float* wqkv = (const float*)d_in[3];
  const float* wout = (const float*)d_in[4];
  const float* bout = (const float*)d_in[5];
  float* y = (float*)d_out;

  // Workspace carve-up (256-B aligned chunks, ~17.8 MB total).
  char* w = (char*)d_ws;
  float*          mu  = (float*)w;           w += 256;
  float*          rs  = (float*)w;           w += 256;
  unsigned short* wqb = (unsigned short*)w;  w += (size_t)HID_ * C_ * 2;   //  96 KB
  unsigned short* xnT = (unsigned short*)w;  w += (size_t)B_ * N_ * C_ * 2; //  4 MB
  unsigned short* qk  = (unsigned short*)w;  w += (size_t)B_ * 256 * N_ * 2;//  8 MB
  unsigned short* vt  = (unsigned short*)w;  w += (size_t)B_ * N_ * C_ * 2; //  4 MB
  float*          S   = (float*)w;           w += (size_t)B_ * 4 * 32 * 32 * 4;
  unsigned short* wef = (unsigned short*)w;  w += (size_t)B_ * C_ * C_ * 2;

  k_gn_stats  <<<B_, 256, 0, stream>>>(x, mu, rs);
  k_cvt_w     <<<(HID_ * C_ + 255) / 256, 256, 0, stream>>>(wqkv, wqb);
  k_norm_t    <<<dim3(N_ / 64, C_ / 32, B_), 256, 0, stream>>>(x, gw, gb, mu, rs, xnT);
  k_qkv_gemm  <<<dim3(N_ / 64, 3, B_), 256, 0, stream>>>(wqb, xnT, qk, vt);
  k_smax_q    <<<dim3(C_, B_), 256, 0, stream>>>(qk);
  k_smax_k    <<<dim3(N_ / 256, 4, B_), 256, 0, stream>>>(qk);
  k_s_gemm    <<<B_ * 4, 128, 0, stream>>>(qk, S);
  k_weff      <<<B_ * 4, 256, 0, stream>>>(wout, S, wef);
  k_final_gemm<<<dim3(N_ / 64, 1, B_), 256, 0, stream>>>(wef, vt, bout, y);
}
